// EvoformerBlock_26783416058539
// MI455X (gfx1250) — matrix operand order repaired and checked by reading, not compiled, not run on hardware
//
#include <hip/hip_runtime.h>
#include <hip/hip_bf16.h>
#include <math.h>

// ---------------------------------------------------------------------------
// Evoformer block for MI455X (gfx1250, wave32, WMMA bf16 -> f32).
// All channel contractions run through v_wmma_f32_16x16x32_bf16.
// All per-kernel offsets fit in 31 bits -> 32-bit index math only.
// ---------------------------------------------------------------------------

typedef __attribute__((ext_vector_type(16))) __bf16 bf16x16;
typedef __attribute__((ext_vector_type(8)))  float  f32x8;

#define DEVI __device__ __forceinline__

DEVI f32x8 wmma_bf16(bf16x16 a, bf16x16 b, f32x8 c) {
  return __builtin_amdgcn_wmma_f32_16x16x32_bf16(false, a, false, b, (short)0, c,
                                                 false, false);
}

// A fragment: 16(M) x 32(K) bf16 tile, row-major in LDS with row stride `ld`.
// lanes 0-15 = rows; elems e -> K=half*16+e.
DEVI bf16x16 load_frag_a(const __bf16* base, int ld, int lane) {
  bf16x16 f;
  int hf = lane >> 4, mr = lane & 15;
  const __bf16* row = base + mr * ld;
  int kb = hf * 16;
#pragma unroll
  for (int e = 0; e < 8; ++e) f[e] = row[kb + e];
#pragma unroll
  for (int e = 0; e < 8; ++e) f[8 + e] = row[kb + 8 + e];
  return f;
}

// B fragment: B[k][n] with the 16 "n" rows (weight rows) stored row-major in LDS
// (each row = 32 K values). lane&15 = n, elems e -> K = (lane>=16?16:0)+e.
DEVI bf16x16 load_frag_b(const __bf16* base, int ld, int lane) {
  bf16x16 f;
  int n = lane & 15, koff = (lane >> 4) * 16;
  const __bf16* row = base + n * ld;
#pragma unroll
  for (int e = 0; e < 16; ++e) f[e] = row[koff + e];
  return f;
}

DEVI float red_max16(float v) {
#pragma unroll
  for (int m = 8; m >= 1; m >>= 1) v = fmaxf(v, __shfl_xor(v, m, 32));
  return v;
}
DEVI float red_sum16(float v) {
#pragma unroll
  for (int m = 8; m >= 1; m >>= 1) v += __shfl_xor(v, m, 32);
  return v;
}
DEVI float red_sum32(float v) {
#pragma unroll
  for (int m = 16; m >= 1; m >>= 1) v += __shfl_xor(v, m, 32);
  return v;
}

DEVI void cvt4(__bf16* dst, float4 f) {
  dst[0] = (__bf16)f.x; dst[1] = (__bf16)f.y;
  dst[2] = (__bf16)f.z; dst[3] = (__bf16)f.w;
}

// ---------------------------------------------------------------------------
// LayerNorm over last dim C. One wave per row. Input row r is read at
// (r/inner)*so + (r%inner)*si (supports the tai transpose view); output plain.
// ---------------------------------------------------------------------------
__global__ void ln_kernel(const float* __restrict__ x, const float* __restrict__ g,
                          const float* __restrict__ b, float* __restrict__ y,
                          int T, int C, int inner, int so, int si) {
  int w    = (int)((blockIdx.x * blockDim.x + threadIdx.x) >> 5);
  int lane = threadIdx.x & 31;
  if (w >= T) return;
  const float* xr = x + (w / inner) * so + (w % inner) * si;
  float s = 0.f, ss = 0.f;
  for (int i = lane; i < C; i += 32) { float v = xr[i]; s += v; ss += v * v; }
  s  = red_sum32(s);
  ss = red_sum32(ss);
  float mu   = s / C;
  float var  = ss / C - mu * mu;
  float rstd = rsqrtf(var + 1e-5f);
  float* yr = y + w * C;
  for (int i = lane; i < C; i += 32) yr[i] = (xr[i] - mu) * rstd * g[i] + b[i];
}

// ---------------------------------------------------------------------------
// Generic NT GEMM: C[r, col] = act( sum_k A[r,k] * W[col,k] + bias[col] ) (+=).
// A row r at (r/a_inner)*a_so + (r%a_inner)*a_si (k contiguous).
// C row r at (r/c_inner)*c_so + (r%c_inner)*c_si (col contiguous).
// Block = 256 thr (8 waves), tile 128(M) x 128(N), K steps of 32.
// Each wave: 32x64 -> 8 WMMAs per K-step. Staging via float4 -> bf16 LDS.
// Requires M%128==0, N%128==0, K%32==0 (true for all uses here).
// ---------------------------------------------------------------------------
template <int ACT, bool ADD>
__global__ __launch_bounds__(256) void gemm_nt_kernel(
    const float* __restrict__ A, int a_inner, int a_so, int a_si,
    const float* __restrict__ W, const float* __restrict__ bias,
    float* __restrict__ C, int c_inner, int c_so, int c_si, int K) {
  __shared__ __bf16 As[128 * 40];
  __shared__ __bf16 Bs[128 * 40];
  const int tid = threadIdx.x;
  const int lane = tid & 31, w = tid >> 5;
  const int m0 = blockIdx.y * 128, n0 = blockIdx.x * 128;
  const int wm = (w >> 1) * 32;   // 0,32,64,96
  const int wn = (w & 1) * 64;    // 0,64
  f32x8 acc[2][4] = {};
  const int srow = tid >> 1, skb = (tid & 1) * 16;  // staging: 16 floats/thread
  for (int k0 = 0; k0 < K; k0 += 32) {
    {  // stage A: 128x32
      int r = m0 + srow;
      const float4* s4 = (const float4*)(A + (r / a_inner) * a_so + (r % a_inner) * a_si + k0 + skb);
      __bf16* dst = &As[srow * 40 + skb];
#pragma unroll
      for (int v = 0; v < 4; ++v) cvt4(dst + v * 4, s4[v]);
    }
    {  // stage B: 128x32 (weight rows)
      const float4* s4 = (const float4*)(W + (n0 + srow) * K + k0 + skb);
      __bf16* dst = &Bs[srow * 40 + skb];
#pragma unroll
      for (int v = 0; v < 4; ++v) cvt4(dst + v * 4, s4[v]);
    }
    __syncthreads();
    bf16x16 af[2], bfr[4];
#pragma unroll
    for (int t = 0; t < 2; ++t) af[t]  = load_frag_a(&As[(wm + t * 16) * 40], 40, lane);
#pragma unroll
    for (int t = 0; t < 4; ++t) bfr[t] = load_frag_b(&Bs[(wn + t * 16) * 40], 40, lane);
#pragma unroll
    for (int tm = 0; tm < 2; ++tm)
#pragma unroll
      for (int tn = 0; tn < 4; ++tn)
        acc[tm][tn] = wmma_bf16(af[tm], bfr[tn], acc[tm][tn]);
    __syncthreads();
  }
  const int hf = lane >> 4, cl = lane & 15;
#pragma unroll
  for (int tm = 0; tm < 2; ++tm)
#pragma unroll
    for (int tn = 0; tn < 4; ++tn) {
      int col  = n0 + wn + tn * 16 + cl;
      float bv = bias ? bias[col] : 0.f;
#pragma unroll
      for (int r = 0; r < 8; ++r) {
        int row = m0 + wm + tm * 16 + r + hf * 8;
        int off = (row / c_inner) * c_so + (row % c_inner) * c_si + col;
        float v = acc[tm][tn][r] + bv;
        if (ACT == 1) v = fmaxf(v, 0.f);
        if (ACT == 2) v = 1.f / (1.f + __expf(-v));
        if (ADD) C[off] += v; else C[off] = v;
      }
    }
}

// ---------------------------------------------------------------------------
// Fused MHA core: one wave per (batch b, head h, 16-row q tile).
// qkv rows: (b*S + t) * E3, layout [q | k | v], D = E/H.
// bias (optional) layout (q, k, h): bias[(q*S + k)*H + h] (pair-bias tables).
// ST = S/16 (even), D in {16, 32} (D=16 zero-pads K of Q.K^T to 32).
// out rows: (b*S + q)*E + h*D + d.
// ---------------------------------------------------------------------------
template <int ST, int D>
__global__ __launch_bounds__(32) void mha_kernel(
    const float* __restrict__ qkv, int S, int E3, int E, int H,
    const float* __restrict__ bias, float scale, float* __restrict__ out) {
  constexpr int SP = ST * 16 + 8;
  __shared__ __bf16 Pl[16 * SP];
  const int lane = threadIdx.x;
  const int q0 = blockIdx.x * 16, h = blockIdx.y, b = blockIdx.z;
  const int hf = lane >> 4, l16 = lane & 15;
  const float* base = qkv + b * S * E3;

  // Q fragment (A layout), zero-padded to K=32 when D==16
  bf16x16 qf;
  {
    const float* qrow = base + (q0 + l16) * E3 + h * D;
    int kb = hf * 16;
#pragma unroll
    for (int e = 0; e < 8; ++e) { int d = kb + e;      qf[e]     = (__bf16)(d < D ? qrow[d] : 0.f); }
#pragma unroll
    for (int e = 0; e < 8; ++e) { int d = kb + 8 + e;  qf[8 + e] = (__bf16)(d < D ? qrow[d] : 0.f); }
  }

  // scores = Q K^T
  f32x8 sc[ST];
#pragma unroll
  for (int jt = 0; jt < ST; ++jt) {
    bf16x16 kf;
    const float* krow = base + (jt * 16 + l16) * E3 + E + h * D;
    int koff = hf * 16;
#pragma unroll
    for (int e = 0; e < 16; ++e) { int d = koff + e; kf[e] = (__bf16)(d < D ? krow[d] : 0.f); }
    f32x8 zz = {};
    sc[jt] = wmma_bf16(qf, kf, zz);
  }

  // scale + bias + row softmax; write P (bf16) to LDS in (q, key) layout
  float rinv[8];
#pragma unroll
  for (int r = 0; r < 8; ++r) {
    int q = q0 + r + hf * 8;
    float mx = -1e30f;
#pragma unroll
    for (int jt = 0; jt < ST; ++jt) {
      float v = sc[jt][r] * scale;
      if (bias) v += bias[(q * S + jt * 16 + l16) * H + h];
      sc[jt][r] = v;
      mx = fmaxf(mx, v);
    }
    mx = red_max16(mx);
    float sum = 0.f;
#pragma unroll
    for (int jt = 0; jt < ST; ++jt) {
      float p = __expf(sc[jt][r] - mx);
      sum += p;
      Pl[(r + hf * 8) * SP + jt * 16 + l16] = (__bf16)p;
    }
    sum = red_sum16(sum);
    rinv[r] = 1.f / sum;
  }

  // O = P @ V
  constexpr int DT = D / 16;
  f32x8 oacc[DT] = {};
#pragma unroll
  for (int kt = 0; kt < ST / 2; ++kt) {
    bf16x16 pf = load_frag_a(&Pl[kt * 32], SP, lane);
    int koff = hf * 16;
#pragma unroll
    for (int dt = 0; dt < DT; ++dt) {
      bf16x16 vf;
#pragma unroll
      for (int e = 0; e < 16; ++e) {
        int key = kt * 32 + koff + e;
        vf[e] = (__bf16)base[key * E3 + 2 * E + h * D + dt * 16 + l16];
      }
      oacc[dt] = wmma_bf16(pf, vf, oacc[dt]);
    }
  }
#pragma unroll
  for (int dt = 0; dt < DT; ++dt)
#pragma unroll
    for (int r = 0; r < 8; ++r) {
      int q = q0 + r + hf * 8;
      out[(b * S + q) * E + h * D + dt * 16 + l16] = oacc[dt][r] * rinv[r];
    }
}

// ---------------------------------------------------------------------------
// Triangle multiplicative einsum: x[i,j,c] = sum_k A[i,k,c] * B[j,k,c]
// (strides select outgoing bikc,bjkc or incoming bkic,bkjc).
// One wave per (16x16 (i,j) tile, channel c). WMMA with strided LDS staging.
// ---------------------------------------------------------------------------
__global__ __launch_bounds__(32) void trimul_kernel(
    const float* __restrict__ a, const float* __restrict__ bmat,
    float* __restrict__ x, int L, int C, int a_rs, int a_ks, int b_rs, int b_ks) {
  __shared__ __bf16 At[16 * 40];
  __shared__ __bf16 Bt[16 * 40];
  const int lane = threadIdx.x;
  const int i0 = blockIdx.x * 16, j0 = blockIdx.y * 16, c = blockIdx.z;
  const int hf = lane >> 4, l16 = lane & 15;
  f32x8 acc = {};
  for (int k0 = 0; k0 < L; k0 += 32) {
    int row = l16, kb = hf * 16;
    const float* pa = a    + (i0 + row) * a_rs + (k0 + kb) * a_ks + c;
    const float* pb = bmat + (j0 + row) * b_rs + (k0 + kb) * b_ks + c;
#pragma unroll
    for (int e = 0; e < 16; ++e) {
      At[row * 40 + kb + e] = (__bf16)pa[e * a_ks];
      Bt[row * 40 + kb + e] = (__bf16)pb[e * b_ks];
    }
    bf16x16 af  = load_frag_a(At, 40, lane);
    bf16x16 bf2 = load_frag_b(Bt, 40, lane);
    acc = wmma_bf16(af, bf2, acc);
  }
#pragma unroll
  for (int r = 0; r < 8; ++r) {
    int i = i0 + r + hf * 8, j = j0 + l16;
    x[(i * L + j) * C + c] = acc[r];
  }
}

// ---------------------------------------------------------------------------
// Small-N projection (pair bias, N=8): Y[r,n] = dot(X[r,:], W[n,:]) (no bias).
// ---------------------------------------------------------------------------
__global__ void gemm_small_kernel(const float* __restrict__ X, const float* __restrict__ W,
                                  float* __restrict__ Y, int T, int K, int N) {
  int t = blockIdx.x * blockDim.x + threadIdx.x;
  if (t >= T * N) return;
  int r = t / N, n = t % N;
  const float* xr = X + r * K;
  const float* wr = W + n * K;
  float s = 0.f;
  for (int k = 0; k < K; ++k) s += xr[k] * wr[k];
  Y[t] = s;
}

// outer-product-mean stage 1: outv[l*8+c] = mean_n dot(mn[(n,l),:], w[c,:]) + b[c]
__global__ void opm_reduce_kernel(const float* __restrict__ mn, const float* __restrict__ w,
                                  const float* __restrict__ b, float* __restrict__ outv,
                                  int N, int L, int C) {
  int wv = (int)((blockIdx.x * blockDim.x + threadIdx.x) >> 5);
  int lane = threadIdx.x & 31;
  if (wv >= L * 8) return;
  int l = wv / 8, c = wv % 8;
  const float* wr = w + c * C;
  float s = 0.f;
  for (int n = 0; n < N; ++n) {
    const float* xr = mn + (n * L + l) * C;
    for (int k = lane; k < C; k += 32) s += xr[k] * wr[k];
  }
  s = red_sum32(s);
  if (lane == 0) outv[wv] = s / N + b[c];
}

// outer-product-mean stage 2: z[i,j,o] += bias[o] + sum_{c,d} W[o,c*8+d]*a[i,c]*b[j,d]
__global__ void opm_outer_kernel(const float* __restrict__ av, const float* __restrict__ bv,
                                 const float* __restrict__ w, const float* __restrict__ bias,
                                 float* __restrict__ z, int L, int C) {
  int t = blockIdx.x * blockDim.x + threadIdx.x;
  if (t >= L * L * C) return;
  int o = t % C;
  int r = t / C;
  int i = r / L, j = r % L;
  float acc = bias[o];
  const float* wr = w + o * 64;
#pragma unroll
  for (int c = 0; c < 8; ++c) {
    float ai = av[i * 8 + c];
#pragma unroll
    for (int d = 0; d < 8; ++d) acc += wr[c * 8 + d] * ai * bv[j * 8 + d];
  }
  z[t] += acc;
}

__global__ void mul_kernel(float* __restrict__ a, const float* __restrict__ g, int n) {
  int t = blockIdx.x * blockDim.x + threadIdx.x;
  if (t < n) a[t] *= g[t];
}

// z[(i,j) or (j,i), ch] += gate[r,ch] * p[r,ch]   (swap for 'tai')
__global__ void gate_addz_kernel(float* __restrict__ z, const float* __restrict__ g,
                                 const float* __restrict__ p, int L, int C, int swap) {
  int t = blockIdx.x * blockDim.x + threadIdx.x;
  if (t >= L * L * C) return;
  int ch = t % C;
  int r = t / C;
  int i = r / L, j = r % L;
  int off = swap ? ((j * L + i) * C + ch) : t;
  z[off] += g[t] * p[t];
}

// ---------------------------------------------------------------------------
// Host side
// ---------------------------------------------------------------------------
namespace {
constexpr int  Nn = 64, Ll = 192, CM = 256, CZ = 128, Hh = 8;
constexpr int  TM = Nn * Ll;                // 12288 MSA tokens
constexpr int  TZ = Ll * Ll;                // 36864 pair tokens
constexpr long NM = (long)TM * CM;          // 3,145,728
constexpr long NZ = (long)TZ * CZ;          // 4,718,592

static void gemm(hipStream_t s, int act, bool add,
                 const float* A, int a_inner, int a_so, int a_si,
                 const float* W, const float* bias,
                 float* C, int c_inner, int c_so, int c_si,
                 int M, int Ncols, int K) {
  dim3 grid(Ncols / 128, M / 128), blk(256);
  if (add) {
    gemm_nt_kernel<0, true><<<grid, blk, 0, s>>>(A, a_inner, a_so, a_si, W, bias, C, c_inner, c_so, c_si, K);
  } else if (act == 1) {
    gemm_nt_kernel<1, false><<<grid, blk, 0, s>>>(A, a_inner, a_so, a_si, W, bias, C, c_inner, c_so, c_si, K);
  } else if (act == 2) {
    gemm_nt_kernel<2, false><<<grid, blk, 0, s>>>(A, a_inner, a_so, a_si, W, bias, C, c_inner, c_so, c_si, K);
  } else {
    gemm_nt_kernel<0, false><<<grid, blk, 0, s>>>(A, a_inner, a_so, a_si, W, bias, C, c_inner, c_so, c_si, K);
  }
}

static void ln(hipStream_t s, const float* x, const float* g, const float* b, float* y,
               int T, int C, int inner, int so, int si) {
  dim3 grid((unsigned)((T * 32 + 255) / 256)), blk(256);
  ln_kernel<<<grid, blk, 0, s>>>(x, g, b, y, T, C, inner, so, si);
}
}  // namespace

extern "C" void kernel_launch(void* const* d_in, const int* in_sizes, int n_in,
                              void* d_out, int out_size, void* d_ws, size_t ws_size,
                              hipStream_t stream) {
  (void)in_sizes; (void)n_in; (void)out_size; (void)ws_size;
  // Parameter flattening: setup_inputs() dict insertion order, recursive.
  //  0 m, 1 z
  //  row:     2 nm_g 3 nm_b 4 nz_g 5 nz_b 6 pb_w 7 in_w 8 in_b 9 out_w 10 out_b
  //  col:     11 n_g 12 n_b 13 in_w 14 in_b 15 out_w 16 out_b
  //  msa_ff:  17 n_g 18 n_b 19 l1w 20 l1b 21 l2w 22 l2b
  //  opm:     23 n_g 24 n_b 25 piw 26 pib 27 pjw 28 pjb 29 outw 30 outb
  //  tmo:     31..46  (n_g n_b n2_g n2_b pa.w pa.b pb.w pb.b ga.w ga.b gb.w gb.b o.w o.b g.w g.b)
  //  tmi:     47..62
  //  tao:     63..71  (n_g n_b pb_w g.w g.b in_w in_b out_w out_b)
  //  tai:     72..80
  //  pair_ff: 81..86  (n_g n_b l1w l1b l2w l2b)
  auto P = [&](int i) { return (const float*)d_in[i]; };

  float* m = (float*)d_out;           // (N,L,CM)
  float* z = m + NM;                  // (L,L,CZ)
  hipMemcpyAsync(m, d_in[0], NM * sizeof(float), hipMemcpyDeviceToDevice, stream);
  hipMemcpyAsync(z, d_in[1], NZ * sizeof(float), hipMemcpyDeviceToDevice, stream);

  // workspace arena (floats)
  float* BUF_A = (float*)d_ws;            // TZ*512 = 18,874,368 (qkv / ff hidden)
  float* BUF_B = BUF_A + (long)TZ * 512;  // NZ   zn
  float* BUF_C = BUF_B + NZ;              // NZ   attn-out / trimul a
  float* BUF_D = BUF_C + NZ;              // NZ   trimul b / tri-attn proj
  float* BUF_E = BUF_D + NZ;              // NZ   trimul x
  float* BUF_F = BUF_E + NZ;              // NZ   sigmoid gates
  float* BUF_G = BUF_F + NZ;              // NZ   o-proj
  float* BUF_H = BUF_G + NZ;              // NM   normalized m
  float* BUF_S = BUF_H + NM;              // TZ*8 pair-bias tables
  float* BUF_V = BUF_S + (long)TZ * 8;    // opm a/b vectors (2 * L*8)

  const float sc32 = 0.1767766952966369f;   // 1/sqrt(32)
  const float sc16 = 0.25f;                 // 1/sqrt(16)
  const int eb = 256;                       // elementwise block
  const unsigned gz = (unsigned)((NZ + eb - 1) / eb);

  // ---- Step 1: MSA row attention with pair bias ------------------------------
  ln(stream, z, P(4), P(5), BUF_B, TZ, CZ, TZ, 0, CZ);
  gemm_small_kernel<<<(unsigned)((TZ * 8 + eb - 1) / eb), eb, 0, stream>>>(BUF_B, P(6), BUF_S, TZ, CZ, 8);
  ln(stream, m, P(2), P(3), BUF_H, TM, CM, TM, 0, CM);
  gemm(stream, 0, false, BUF_H, TM, 0, CM, P(7), P(8), BUF_A, TM, 0, 768, TM, 768, CM);
  mha_kernel<12, 32><<<dim3(12, 8, 64), 32, 0, stream>>>(BUF_A, Ll, 768, CM, Hh, BUF_S, sc32, BUF_C);
  gemm(stream, 0, true, BUF_C, TM, 0, CM, P(9), P(10), m, TM, 0, CM, TM, CM, CM);

  // ---- Step 2: MSA column attention ------------------------------------------
  ln(stream, m, P(11), P(12), BUF_H, TM, CM, TM, 0, CM);
  // rows ordered (l, n): A offset = l*CM + n*(L*CM)
  gemm(stream, 0, false, BUF_H, Nn, CM, Ll * CM, P(13), P(14), BUF_A, TM, 0, 768, TM, 768, CM);
  mha_kernel<4, 32><<<dim3(4, 8, 192), 32, 0, stream>>>(BUF_A, Nn, 768, CM, Hh, nullptr, sc32, BUF_C);
  gemm(stream, 0, true, BUF_C, TM, 0, CM, P(15), P(16), m, Nn, CM, Ll * CM, TM, CM, CM);

  // ---- Step 3: MSA transition -------------------------------------------------
  ln(stream, m, P(17), P(18), BUF_H, TM, CM, TM, 0, CM);
  gemm(stream, 1, false, BUF_H, TM, 0, CM, P(19), P(20), BUF_A, TM, 0, 512, TM, 512, CM);
  gemm(stream, 0, true, BUF_A, TM, 0, 512, P(21), P(22), m, TM, 0, CM, TM, CM, 512);

  // ---- Step 4: outer product mean --------------------------------------------
  ln(stream, m, P(23), P(24), BUF_H, TM, CM, TM, 0, CM);
  opm_reduce_kernel<<<(Ll * 8 * 32 + eb - 1) / eb, eb, 0, stream>>>(BUF_H, P(25), P(26), BUF_V, Nn, Ll, CM);
  opm_reduce_kernel<<<(Ll * 8 * 32 + eb - 1) / eb, eb, 0, stream>>>(BUF_H, P(27), P(28), BUF_V + 2048, Nn, Ll, CM);
  opm_outer_kernel<<<gz, eb, 0, stream>>>(BUF_V, BUF_V + 2048, P(29), P(30), z, Ll, CZ);

  // ---- Steps 5-6: triangle multiplicative updates ----------------------------
  for (int u = 0; u < 2; ++u) {
    const int base = (u == 0) ? 31 : 47;
    const bool og = (u == 0);
    ln(stream, z, P(base + 0), P(base + 1), BUF_B, TZ, CZ, TZ, 0, CZ);
    gemm(stream, 2, false, BUF_B, TZ, 0, CZ, P(base + 8),  P(base + 9),  BUF_F, TZ, 0, CZ, TZ, CZ, CZ); // sig(ga)
    gemm(stream, 0, false, BUF_B, TZ, 0, CZ, P(base + 4),  P(base + 5),  BUF_C, TZ, 0, CZ, TZ, CZ, CZ); // pa
    mul_kernel<<<gz, eb, 0, stream>>>(BUF_C, BUF_F, (int)NZ);                                           // a
    gemm(stream, 2, false, BUF_B, TZ, 0, CZ, P(base + 10), P(base + 11), BUF_F, TZ, 0, CZ, TZ, CZ, CZ); // sig(gb)
    gemm(stream, 0, false, BUF_B, TZ, 0, CZ, P(base + 6),  P(base + 7),  BUF_D, TZ, 0, CZ, TZ, CZ, CZ); // pb
    mul_kernel<<<gz, eb, 0, stream>>>(BUF_D, BUF_F, (int)NZ);                                           // b
    int rs = og ? Ll * CZ : CZ;
    int ks = og ? CZ : Ll * CZ;
    trimul_kernel<<<dim3(12, 12, 128), 32, 0, stream>>>(BUF_C, BUF_D, BUF_E, Ll, CZ, rs, ks, rs, ks);
    ln(stream, BUF_E, P(base + 2), P(base + 3), BUF_E, TZ, CZ, TZ, 0, CZ);                              // ln(x)
    gemm(stream, 2, false, BUF_B, TZ, 0, CZ, P(base + 14), P(base + 15), BUF_F, TZ, 0, CZ, TZ, CZ, CZ); // sig(g)
    gemm(stream, 0, false, BUF_E, TZ, 0, CZ, P(base + 12), P(base + 13), BUF_G, TZ, 0, CZ, TZ, CZ, CZ); // o
    gate_addz_kernel<<<gz, eb, 0, stream>>>(z, BUF_F, BUF_G, Ll, CZ, 0);
  }

  // ---- Steps 7-8: triangle attention -----------------------------------------
  for (int u = 0; u < 2; ++u) {
    const int base = (u == 0) ? 63 : 72;
    const bool og = (u == 0);
    if (og) ln(stream, z, P(base), P(base + 1), BUF_B, TZ, CZ, TZ, 0, CZ);
    else    ln(stream, z, P(base), P(base + 1), BUF_B, TZ, CZ, Ll, CZ, Ll * CZ);  // transposed view
    gemm_small_kernel<<<(unsigned)((TZ * 8 + eb - 1) / eb), eb, 0, stream>>>(BUF_B, P(base + 2), BUF_S, TZ, CZ, 8);
    gemm(stream, 2, false, BUF_B, TZ, 0, CZ, P(base + 3), P(base + 4), BUF_F, TZ, 0, CZ, TZ, CZ, CZ);   // gate
    gemm(stream, 0, false, BUF_B, TZ, 0, CZ, P(base + 5), P(base + 6), BUF_A, TZ, 0, 384, TZ, 384, CZ); // qkv
    mha_kernel<12, 16><<<dim3(12, 8, 192), 32, 0, stream>>>(BUF_A, Ll, 384, CZ, Hh, BUF_S, sc16, BUF_C);
    gemm(stream, 0, false, BUF_C, TZ, 0, CZ, P(base + 7), P(base + 8), BUF_D, TZ, 0, CZ, TZ, CZ, CZ);   // proj
    gate_addz_kernel<<<gz, eb, 0, stream>>>(z, BUF_F, BUF_D, Ll, CZ, og ? 0 : 1);
  }

  // ---- Step 9: pair transition ------------------------------------------------
  ln(stream, z, P(81), P(82), BUF_B, TZ, CZ, TZ, 0, CZ);
  gemm(stream, 1, false, BUF_B, TZ, 0, CZ, P(83), P(84), BUF_A, TZ, 0, 512, TZ, 512, CZ);
  gemm(stream, 0, true, BUF_A, TZ, 0, 512, P(85), P(86), z, TZ, 0, CZ, TZ, CZ, 512);
}
